// DeepseekV3DecoderLayer_35485019799864
// MI455X (gfx1250) — compile-verified, matrix operand order checked by reading, not run on hardware
//
#include <hip/hip_runtime.h>
#include <hip/hip_bf16.h>

// ---------------- constants ----------------
constexpr int T    = 1024;
constexpr int Hd   = 1024;   // hidden
constexpr int NH   = 8;
constexpr int NKV  = 4;
constexpr int HDim = 128;    // head dim
constexpr int KVD  = NKV * HDim;  // 512
constexpr int NE   = 32;     // experts
constexpr int TOPK = 4;
constexpr int NG   = 8;
constexpr int TKG  = 4;
constexpr int MI   = 512;
constexpr int SI   = 1024;
constexpr int ASSIGN = T * TOPK; // 4096 expert-token assignments

typedef __bf16 bf16;
typedef __attribute__((ext_vector_type(16))) __bf16 bf16x16;
typedef __attribute__((ext_vector_type(8)))  float  f32x8;

static __device__ __forceinline__ bf16 f2bf(float f) {
  union { float f; unsigned u; } v; v.f = f;
  unsigned r = v.u + 0x7FFFu + ((v.u >> 16) & 1u);  // RNE
  unsigned short h = (unsigned short)(r >> 16);
  return __builtin_bit_cast(bf16, h);
}

enum { MODE_PLAIN = 0, MODE_RESID = 1, MODE_CAUSAL = 2 };

// ---------------- generic WMMA GEMM ----------------
// C[M,N] = A[M,K](bf16) * B(fp32 -> bf16 staged in LDS)
// bKN==1: B row-major KxN ; bKN==0: B row-major NxK (op(B)=B^T)
// Block tile 128x64x32, 4 waves, each wave 64x32 -> 8 v_wmma per k-step.
// Fast path: double-buffered LDS; A staged via global_load_async_to_lds_b128
// (ASYNCcnt), B loaded to regs ahead of the MMAs, drained with s_wait_asynccnt.
#define BM 128
#define BN 64
#define BK 32
#define BKP (BK + 8)   // padded B row stride (40 bf16 = 80B = 20 banks, conflict-free)

__global__ __launch_bounds__(128) void wmma_gemm_kernel(
    const bf16* __restrict__ A, int lda, long long sAz,
    const float* __restrict__ B, int ldb, long long sBz, int bZdiv, int bKN,
    float* __restrict__ C, int ldc, long long sCz,
    int M, int N, int K,
    int mode, const float* __restrict__ resid, int ldres, float scale,
    const int* __restrict__ mCounts,   // per-z M override (experts)
    const int* __restrict__ zRowOfs,   // per-z base into compacted row lists
    const int* __restrict__ aRowIdx,   // gather list for A rows
    const int* __restrict__ cRowIdx)   // scatter list for C rows
{
  const int z  = blockIdx.z;
  const int Mz = mCounts ? mCounts[z] : M;
  const int m0 = blockIdx.y * BM;
  if (m0 >= Mz) return;
  const int n0 = blockIdx.x * BN;
  if (mode == MODE_CAUSAL && (m0 + BM) <= n0) return;  // fully masked tile

  const long long L = zRowOfs ? (long long)zRowOfs[z] : 0;
  const bf16*  Az = A + (size_t)z * (size_t)sAz;
  const float* Bz = B + (size_t)(z / bZdiv) * (size_t)sBz;
  float*       Cz = C + (size_t)z * (size_t)sCz;

  __shared__ bf16 At[2][BM * BK];   // 2 x 8KB, 64B rows + XOR chunk swizzle
  __shared__ bf16 Bt[2][BN][BKP];   // 2 x 5KB, [n][k] for BOTH B layouts

  const int tid  = threadIdx.x;
  const int lane = tid & 31;
  const int wave = tid >> 5;
  const int wm   = wave >> 1;   // 0..1 -> 64-row strip
  const int wn   = wave & 1;    // 0..1 -> 32-col strip
  const int half = lane >> 4;
  const int rr   = lane & 15;

  f32x8 acc[4][2];
  #pragma unroll
  for (int im = 0; im < 4; ++im)
    #pragma unroll
    for (int in = 0; in < 2; ++in)
      #pragma unroll
      for (int i = 0; i < 8; ++i) acc[im][in][i] = 0.f;

  const int nK = (K + BK - 1) / BK;

  // ---- helpers ----
  auto aRowOf = [&](int m) -> long long {
    return aRowIdx ? (long long)aRowIdx[L + m]
                   : (zRowOfs ? (L + m) : (long long)m);
  };
  // async DMA A tile kb -> LDS buffer buf (4 x b128 per thread)
  auto stageA_async = [&](int kb, int buf) {
    const int k0 = kb * BK;
    #pragma unroll
    for (int it = 0; it < 4; ++it) {
      const int cid = tid + it * 128;        // 0..511 16B chunks
      const int r   = cid >> 2;              // 0..127
      const int cch = cid & 3;
      const long long ar = aRowOf(m0 + r);
      const int pk  = ((cch ^ (r & 3)) << 3);
      unsigned ldsa = (unsigned)(uintptr_t)&At[buf][r * BK + pk];
      unsigned long long ga =
          (unsigned long long)(uintptr_t)(Az + ar * (long long)lda + k0 + cch * 8);
      asm volatile("global_load_async_to_lds_b128 %0, %1, off"
                   :: "v"(ldsa), "v"(ga) : "memory");
    }
  };
  // B tile kb -> registers (fp32), 16 values per thread
  auto loadB_regs = [&](int kb, float4* regs) {
    const int k0 = kb * BK;
    if (!bKN) {       // NxK: contiguous k
      #pragma unroll
      for (int it = 0; it < 4; ++it) {
        const int j  = tid + it * 128;
        const int n  = j >> 3;
        const int kc = (j & 7) << 2;
        regs[it] = *(const float4*)&Bz[(long long)(n0 + n) * ldb + k0 + kc];
      }
    } else {          // KxN: 4 strided dwords, coalesced across lanes
      #pragma unroll
      for (int it = 0; it < 4; ++it) {
        const int j  = tid + it * 128;
        const int n  = j & 63;
        const int kc = (j >> 6) << 2;
        const float* src = &Bz[(long long)(k0 + kc) * ldb + n0 + n];
        regs[it].x = src[0];
        regs[it].y = src[(long long)ldb];
        regs[it].z = src[(long long)ldb * 2];
        regs[it].w = src[(long long)ldb * 3];
      }
    }
  };
  auto storeB_lds = [&](int buf, const float4* regs) {
    #pragma unroll
    for (int it = 0; it < 4; ++it) {
      const int j  = tid + it * 128;
      const int n  = bKN ? (j & 63) : (j >> 3);
      const int kc = bKN ? ((j >> 6) << 2) : ((j & 7) << 2);
      bf16* d = &Bt[buf][n][kc];
      d[0] = f2bf(regs[it].x); d[1] = f2bf(regs[it].y);
      d[2] = f2bf(regs[it].z); d[3] = f2bf(regs[it].w);
    }
  };
  // 8 WMMAs from LDS buffer buf
  auto mma_step = [&](int buf) {
    bf16x16 af[4], bfr[2];
    #pragma unroll
    for (int im = 0; im < 4; ++im) {
      const int ar = wm * 64 + im * 16 + rr;
      const int x0 = ((half       ^ (ar & 3)) << 3);
      const int x1 = (((2 + half) ^ (ar & 3)) << 3);
      #pragma unroll
      for (int jj = 0; jj < 8; ++jj) {
        af[im][jj]     = At[buf][ar * BK + x0 + jj];
        af[im][jj + 8] = At[buf][ar * BK + x1 + jj];
      }
    }
    #pragma unroll
    for (int in = 0; in < 2; ++in) {
      const int bc = wn * 32 + in * 16 + rr;
      #pragma unroll
      for (int jj = 0; jj < 8; ++jj) {
        bfr[in][jj]     = Bt[buf][bc][half * 8 + jj];
        bfr[in][jj + 8] = Bt[buf][bc][16 + half * 8 + jj];
      }
    }
    #pragma unroll
    for (int im = 0; im < 4; ++im)
      #pragma unroll
      for (int in = 0; in < 2; ++in)
        acc[im][in] = __builtin_amdgcn_wmma_f32_16x16x32_bf16(
            false, af[im], false, bfr[in], (short)0, acc[im][in], false, false);
  };

  const bool fastPath = ((m0 + BM) <= Mz) && ((n0 + BN) <= N) && ((K & (BK - 1)) == 0);

  if (fastPath) {
    // ---------------- pipelined fast loop ----------------
    float4 bregs[4];
    stageA_async(0, 0);
    loadB_regs(0, bregs);
    storeB_lds(0, bregs);
    asm volatile("s_wait_asynccnt 0x0" ::: "memory");
    __syncthreads();
    for (int kb = 0; kb < nK; ++kb) {
      const int cur = kb & 1, nxt = cur ^ 1;
      const bool more = (kb + 1) < nK;
      if (more) {
        stageA_async(kb + 1, nxt);   // DMA overlaps the MMAs below
        loadB_regs(kb + 1, bregs);   // VMEM overlaps the MMAs below
        if (kb + 2 < nK)
          __builtin_prefetch(bKN ? &Bz[(long long)(kb + 2) * BK * ldb + n0]
                                 : &Bz[(long long)n0 * ldb + (kb + 2) * BK], 0, 1);
      }
      mma_step(cur);
      if (more) storeB_lds(nxt, bregs);
      asm volatile("s_wait_asynccnt 0x0" ::: "memory");
      __syncthreads();
    }
  } else {
    // ---------------- guarded generic loop (edge tiles only) ----------------
    for (int kb = 0; kb < nK; ++kb) {
      const int k0 = kb * BK;
      for (int i = tid; i < BM * BK; i += 128) {
        const int r = i >> 5, c = i & 31;
        const int m = m0 + r;
        bf16 v = f2bf(0.0f);
        if (m < Mz && (k0 + c) < K)
          v = Az[aRowOf(m) * (long long)lda + (k0 + c)];
        At[0][r * BK + (((c >> 3) ^ (r & 3)) << 3) + (c & 7)] = v;
      }
      for (int i = tid; i < BN * BK; i += 128) {
        const int n = i >> 5, c = i & 31;
        const int gn = n0 + n, gk = k0 + c;
        float f = 0.0f;
        if (gn < N && gk < K)
          f = bKN ? Bz[(long long)gk * ldb + gn]
                  : Bz[(long long)gn * ldb + gk];
        Bt[0][n][c] = f2bf(f);
      }
      __syncthreads();
      mma_step(0);
      __syncthreads();
    }
  }

  // ---- epilogue: D layout -> VGPR i: M=i (lanes 0-15) / M=i+8 (lanes 16-31); N=lane&15 ----
  const int colLo = lane & 15;
  const int rAdd  = (lane >> 4) * 8;
  #pragma unroll
  for (int im = 0; im < 4; ++im) {
    #pragma unroll
    for (int in = 0; in < 2; ++in) {
      const int gn = n0 + wn * 32 + in * 16 + colLo;
      if (gn >= N) continue;
      const int mb = m0 + wm * 64 + im * 16 + rAdd;
      #pragma unroll
      for (int i = 0; i < 8; ++i) {
        const int m = mb + i;
        if (m >= Mz) continue;
        long long orow = cRowIdx ? (long long)cRowIdx[L + m]
                                 : (zRowOfs ? (L + m) : (long long)m);
        float v = acc[im][in][i];
        if (mode == MODE_CAUSAL)
          v = (m >= gn) ? v * scale : -3.4028234663852886e38f;
        else if (mode == MODE_RESID)
          v += resid[(long long)m * ldres + gn];
        Cz[orow * (long long)ldc + gn] = v;
      }
    }
  }
}

// ---------------- elementwise / reduction kernels ----------------
__global__ void rmsnorm_bf16_kernel(const float* __restrict__ x,
                                    const float* __restrict__ w,
                                    bf16* __restrict__ y, int ncols) {
  const int t = blockIdx.x;
  const float* row = x + (long long)t * ncols;
  __shared__ float red[256];
  float s = 0.0f;
  for (int c = threadIdx.x; c < ncols; c += 256) { float v = row[c]; s += v * v; }
  red[threadIdx.x] = s; __syncthreads();
  for (int st = 128; st > 0; st >>= 1) {
    if (threadIdx.x < st) red[threadIdx.x] += red[threadIdx.x + st];
    __syncthreads();
  }
  const float inv = rsqrtf(red[0] / (float)ncols + 1e-6f);
  for (int c = threadIdx.x; c < ncols; c += 256)
    y[(long long)t * ncols + c] = f2bf(row[c] * inv * w[c]);
}

__global__ void rope_kernel(const float* __restrict__ q, const float* __restrict__ k,
                            const int* __restrict__ pos,
                            bf16* __restrict__ qb, float* __restrict__ ko) {
  const int t = blockIdx.x;
  const float p = (float)pos[t];
  for (int i = threadIdx.x; i < Hd + KVD; i += 256) {
    const bool isq = i < Hd;
    const int idx  = isq ? i : (i - Hd);
    const int hd   = idx & (HDim - 1);
    const int head = idx >> 7;
    const int d    = (hd < 64) ? hd : (hd - 64);
    const float fr = __expf(-(float)d * 0.14391156f);  // ln(10000)/64
    const float c  = __cosf(p * fr), sn = __sinf(p * fr);
    const float* src = isq ? (q + (long long)t * Hd + head * HDim)
                           : (k + (long long)t * KVD + head * HDim);
    float out;
    if (hd < 64) out = src[hd] * c - src[hd + 64] * sn;
    else         out = src[hd] * c + src[hd - 64] * sn;
    if (isq) qb[(long long)t * Hd + idx] = f2bf(out);
    else     ko[(long long)t * KVD + idx] = out;
  }
}

__global__ void softmax_bf16_kernel(const float* __restrict__ sc, bf16* __restrict__ P) {
  const int h = blockIdx.y, r = blockIdx.x;
  const float* row = sc + ((long long)h * T + r) * T;
  bf16* prow = P + ((long long)h * T + r) * T;
  __shared__ float red[256];
  float mx = -3.4e38f;
  for (int c = threadIdx.x; c < T; c += 256) mx = fmaxf(mx, row[c]);
  red[threadIdx.x] = mx; __syncthreads();
  for (int st = 128; st > 0; st >>= 1) {
    if (threadIdx.x < st) red[threadIdx.x] = fmaxf(red[threadIdx.x], red[threadIdx.x + st]);
    __syncthreads();
  }
  mx = red[0]; __syncthreads();
  float sum = 0.0f;
  for (int c = threadIdx.x; c < T; c += 256) sum += __expf(row[c] - mx);
  red[threadIdx.x] = sum; __syncthreads();
  for (int st = 128; st > 0; st >>= 1) {
    if (threadIdx.x < st) red[threadIdx.x] += red[threadIdx.x + st];
    __syncthreads();
  }
  const float inv = 1.0f / red[0];
  for (int c = threadIdx.x; c < T; c += 256)
    prow[c] = f2bf(__expf(row[c] - mx) * inv);
}

__global__ void cast_bf16_kernel(const float* __restrict__ in, bf16* __restrict__ out, long long n) {
  long long i = (long long)blockIdx.x * 256 + threadIdx.x;
  if (i < n) out[i] = f2bf(in[i]);
}

__global__ void silu_mul_bf16_kernel(const float* __restrict__ g, const float* __restrict__ u,
                                     bf16* __restrict__ out, long long n) {
  long long i = (long long)blockIdx.x * 256 + threadIdx.x;
  if (i < n) {
    const float gv = g[i];
    out[i] = f2bf((gv / (1.0f + __expf(-gv))) * u[i]);
  }
}

__global__ void router_select_kernel(const float* __restrict__ logits,
                                     const float* __restrict__ bias,
                                     int* __restrict__ tkIdx, float* __restrict__ tkW) {
  const int t = blockIdx.x * blockDim.x + threadIdx.x;
  if (t >= T) return;
  float s[NE], sfc[NE];
  for (int e = 0; e < NE; ++e) {
    const float l = logits[(long long)t * NE + e];
    s[e] = 1.0f / (1.0f + __expf(-l));
    sfc[e] = s[e] + bias[e];
  }
  float gsc[NG];
  for (int g = 0; g < NG; ++g) {
    float hi = -3.4e38f, hi2 = -3.4e38f;
    for (int j = 0; j < NE / NG; ++j) {
      const float v = sfc[g * (NE / NG) + j];
      if (v > hi) { hi2 = hi; hi = v; } else if (v > hi2) hi2 = v;
    }
    gsc[g] = hi + hi2;
  }
  bool gsel[NG];
  for (int g = 0; g < NG; ++g) gsel[g] = false;
  for (int it = 0; it < TKG; ++it) {
    float best = -3.4e38f; int bi = 0;
    for (int g = 0; g < NG; ++g)
      if (!gsel[g] && gsc[g] > best) { best = gsc[g]; bi = g; }
    gsel[bi] = true;
  }
  float mval[NE];
  for (int e = 0; e < NE; ++e) mval[e] = gsel[e >> 2] ? sfc[e] : 0.0f;
  int idx[TOPK]; float w[TOPK]; float wsum = 0.0f;
  for (int it = 0; it < TOPK; ++it) {
    float best = -3.4e38f; int bi = 0;
    for (int e = 0; e < NE; ++e)
      if (mval[e] > best) { best = mval[e]; bi = e; }
    mval[bi] = -3.4e38f;
    idx[it] = bi; w[it] = s[bi]; wsum += s[bi];
  }
  const float inv = 2.5f / (wsum + 1e-20f);
  for (int it = 0; it < TOPK; ++it) {
    tkIdx[(long long)t * TOPK + it] = idx[it];
    tkW[(long long)t * TOPK + it]   = w[it] * inv;
  }
}

// deterministic per-expert token lists (no atomics): 1 block, 32 threads
__global__ void build_lists_kernel(const int* __restrict__ tkIdx,
                                   int* __restrict__ counts, int* __restrict__ offsets,
                                   int* __restrict__ tokens, int* __restrict__ slotmap) {
  const int e = threadIdx.x;
  int c = 0;
  for (int t = 0; t < T; ++t)
    for (int s = 0; s < TOPK; ++s)
      if (tkIdx[t * TOPK + s] == e) ++c;
  counts[e] = c;
  __syncthreads();
  if (e == 0) {
    int o = 0;
    for (int i = 0; i < NE; ++i) { offsets[i] = o; o += counts[i]; }
    offsets[NE] = o;
  }
  __syncthreads();
  int o = offsets[e];
  for (int t = 0; t < T; ++t)
    for (int s = 0; s < TOPK; ++s)
      if (tkIdx[t * TOPK + s] == e) { tokens[o] = t; slotmap[o] = t * TOPK + s; ++o; }
}

__global__ void combine_kernel(const float* __restrict__ shres,
                               const float* __restrict__ slotbuf,
                               const float* __restrict__ tkW,
                               float* __restrict__ out) {
  const long long i = (long long)blockIdx.x * 256 + threadIdx.x;
  if (i >= (long long)T * Hd) return;
  const int t = (int)(i >> 10);
  const int h = (int)(i & (Hd - 1));
  float v = shres[i];
  #pragma unroll
  for (int s = 0; s < TOPK; ++s)
    v += tkW[t * TOPK + s] * slotbuf[(long long)(t * TOPK + s) * Hd + h];
  out[i] = v;
}

// ---------------- host launch ----------------
extern "C" void kernel_launch(void* const* d_in, const int* in_sizes, int n_in,
                              void* d_out, int out_size, void* d_ws, size_t ws_size,
                              hipStream_t stream) {
  (void)in_sizes; (void)n_in; (void)out_size; (void)ws_size;
  const float* hid   = (const float*)d_in[0];
  const int*   pos   = (const int*)  d_in[1];
  const float* ln1   = (const float*)d_in[2];
  const float* ln2   = (const float*)d_in[3];
  const float* q_w   = (const float*)d_in[4];
  const float* k_w   = (const float*)d_in[5];
  const float* v_w   = (const float*)d_in[6];
  const float* o_w   = (const float*)d_in[7];
  const float* rt_w  = (const float*)d_in[8];
  const float* rt_b  = (const float*)d_in[9];
  const float* eg_w  = (const float*)d_in[10];
  const float* eu_w  = (const float*)d_in[11];
  const float* ed_w  = (const float*)d_in[12];
  const float* sg_w  = (const float*)d_in[13];
  const float* su_w  = (const float*)d_in[14];
  const float* sd_w  = (const float*)d_in[15];
  float* out = (float*)d_out;

  char* wsb = (char*)d_ws;
  size_t cur = 0;
  auto alloc = [&](size_t bytes) -> void* {
    void* p = wsb + cur;
    cur += (bytes + 255) & ~(size_t)255;
    return p;
  };
  bf16*  x1b    = (bf16*) alloc((size_t)T * Hd * 2);
  float* qf     = (float*)alloc((size_t)T * Hd * 4);
  float* kf     = (float*)alloc((size_t)T * KVD * 4);
  float* vf     = (float*)alloc((size_t)T * KVD * 4);
  bf16*  qb     = (bf16*) alloc((size_t)T * Hd * 2);
  float* kr     = (float*)alloc((size_t)T * KVD * 4);
  float* sc     = (float*)alloc((size_t)NH * T * T * 4);
  bf16*  Pb     = (bf16*) alloc((size_t)NH * T * T * 2);
  float* attn   = (float*)alloc((size_t)T * Hd * 4);
  bf16*  attnb  = (bf16*) alloc((size_t)T * Hd * 2);
  float* h1     = (float*)alloc((size_t)T * Hd * 4);
  bf16*  x2b    = (bf16*) alloc((size_t)T * Hd * 2);
  float* logits = (float*)alloc((size_t)T * NE * 4);
  int*   tkIdx  = (int*)  alloc((size_t)T * TOPK * 4);
  float* tkW    = (float*)alloc((size_t)T * TOPK * 4);
  int*   counts = (int*)  alloc((size_t)NE * 4);
  int*   offs   = (int*)  alloc((size_t)(NE + 1) * 4);
  int*   tokens = (int*)  alloc((size_t)ASSIGN * 4);
  int*   slotmp = (int*)  alloc((size_t)ASSIGN * 4);
  float* ge     = (float*)alloc((size_t)ASSIGN * MI * 4);
  float* ue     = (float*)alloc((size_t)ASSIGN * MI * 4);
  bf16*  acte   = (bf16*) alloc((size_t)ASSIGN * MI * 2);
  float* slotbf = (float*)alloc((size_t)ASSIGN * Hd * 4);
  float* gs     = (float*)alloc((size_t)T * SI * 4);
  float* us     = (float*)alloc((size_t)T * SI * 4);
  bf16*  acts   = (bf16*) alloc((size_t)T * SI * 2);
  float* shres  = (float*)alloc((size_t)T * Hd * 4);

  const dim3 blk128(128), blk256(256);
  const float attnScale = 0.08838834764831845f;  // 1/sqrt(128)
  auto cdiv = [](int a, int b) { return (a + b - 1) / b; };

  // 1) rmsnorm1 -> bf16
  rmsnorm_bf16_kernel<<<T, blk256, 0, stream>>>(hid, ln1, x1b, Hd);
  // 2) q/k/v projections (weights (out,in) -> B^T flavor)
  wmma_gemm_kernel<<<dim3(cdiv(Hd,BN), cdiv(T,BM), 1), blk128, 0, stream>>>(
      x1b, Hd, 0, q_w, Hd, 0, 1, 0, qf, Hd, 0, T, Hd, Hd,
      MODE_PLAIN, nullptr, 0, 0.f, nullptr, nullptr, nullptr, nullptr);
  wmma_gemm_kernel<<<dim3(cdiv(KVD,BN), cdiv(T,BM), 1), blk128, 0, stream>>>(
      x1b, Hd, 0, k_w, Hd, 0, 1, 0, kf, KVD, 0, T, KVD, Hd,
      MODE_PLAIN, nullptr, 0, 0.f, nullptr, nullptr, nullptr, nullptr);
  wmma_gemm_kernel<<<dim3(cdiv(KVD,BN), cdiv(T,BM), 1), blk128, 0, stream>>>(
      x1b, Hd, 0, v_w, Hd, 0, 1, 0, vf, KVD, 0, T, KVD, Hd,
      MODE_PLAIN, nullptr, 0, 0.f, nullptr, nullptr, nullptr, nullptr);
  // 3) RoPE: q -> bf16, k -> f32
  rope_kernel<<<T, blk256, 0, stream>>>(qf, kf, pos, qb, kr);
  // 4) scores = Q K^T * scale + causal mask (8 heads; GQA kv = h/2)
  wmma_gemm_kernel<<<dim3(cdiv(T,BN), cdiv(T,BM), NH), blk128, 0, stream>>>(
      qb, Hd, HDim, kr, KVD, HDim, 2, 0, sc, T, (long long)T * T, T, T, HDim,
      MODE_CAUSAL, nullptr, 0, attnScale, nullptr, nullptr, nullptr, nullptr);
  // 5) softmax -> bf16 probs
  softmax_bf16_kernel<<<dim3(T, NH), blk256, 0, stream>>>(sc, Pb);
  // 6) out = P @ V  (B = V, KxN flavor)
  wmma_gemm_kernel<<<dim3(cdiv(HDim,BN), cdiv(T,BM), NH), blk128, 0, stream>>>(
      Pb, T, (long long)T * T, vf, KVD, HDim, 2, 1, attn, Hd, HDim, T, HDim, T,
      MODE_PLAIN, nullptr, 0, 0.f, nullptr, nullptr, nullptr, nullptr);
  // 7) o-proj + residual
  cast_bf16_kernel<<<(T * Hd) / 256, blk256, 0, stream>>>(attn, attnb, (long long)T * Hd);
  wmma_gemm_kernel<<<dim3(cdiv(Hd,BN), cdiv(T,BM), 1), blk128, 0, stream>>>(
      attnb, Hd, 0, o_w, Hd, 0, 1, 0, h1, Hd, 0, T, Hd, Hd,
      MODE_RESID, hid, Hd, 0.f, nullptr, nullptr, nullptr, nullptr);
  // 8) rmsnorm2 -> bf16
  rmsnorm_bf16_kernel<<<T, blk256, 0, stream>>>(h1, ln2, x2b, Hd);
  // 9) router logits (N=32, edge path)
  wmma_gemm_kernel<<<dim3(1, cdiv(T,BM), 1), blk128, 0, stream>>>(
      x2b, Hd, 0, rt_w, Hd, 0, 1, 0, logits, NE, 0, T, NE, Hd,
      MODE_PLAIN, nullptr, 0, 0.f, nullptr, nullptr, nullptr, nullptr);
  router_select_kernel<<<T / 256, blk256, 0, stream>>>(logits, rt_b, tkIdx, tkW);
  build_lists_kernel<<<1, 32, 0, stream>>>(tkIdx, counts, offs, tokens, slotmp);
  // 10) routed experts: gathered gate/up GEMMs (eg/eu (H,MI) per expert, KxN)
  wmma_gemm_kernel<<<dim3(cdiv(MI,BN), cdiv(T,BM), NE), blk128, 0, stream>>>(
      x2b, Hd, 0, eg_w, MI, (long long)Hd * MI, 1, 1, ge, MI, 0, T, MI, Hd,
      MODE_PLAIN, nullptr, 0, 0.f, counts, offs, tokens, nullptr);
  wmma_gemm_kernel<<<dim3(cdiv(MI,BN), cdiv(T,BM), NE), blk128, 0, stream>>>(
      x2b, Hd, 0, eu_w, MI, (long long)Hd * MI, 1, 1, ue, MI, 0, T, MI, Hd,
      MODE_PLAIN, nullptr, 0, 0.f, counts, offs, tokens, nullptr);
  silu_mul_bf16_kernel<<<(ASSIGN * MI) / 256, blk256, 0, stream>>>(
      ge, ue, acte, (long long)ASSIGN * MI);
  // 11) expert down-proj, deterministic scatter into (token,slot) buffer
  wmma_gemm_kernel<<<dim3(cdiv(Hd,BN), cdiv(T,BM), NE), blk128, 0, stream>>>(
      acte, MI, 0, ed_w, Hd, (long long)MI * Hd, 1, 1, slotbf, Hd, 0, T, Hd, MI,
      MODE_PLAIN, nullptr, 0, 0.f, counts, offs, nullptr, slotmp);
  // 12) shared expert
  wmma_gemm_kernel<<<dim3(cdiv(SI,BN), cdiv(T,BM), 1), blk128, 0, stream>>>(
      x2b, Hd, 0, sg_w, SI, 0, 1, 1, gs, SI, 0, T, SI, Hd,
      MODE_PLAIN, nullptr, 0, 0.f, nullptr, nullptr, nullptr, nullptr);
  wmma_gemm_kernel<<<dim3(cdiv(SI,BN), cdiv(T,BM), 1), blk128, 0, stream>>>(
      x2b, Hd, 0, su_w, SI, 0, 1, 1, us, SI, 0, T, SI, Hd,
      MODE_PLAIN, nullptr, 0, 0.f, nullptr, nullptr, nullptr, nullptr);
  silu_mul_bf16_kernel<<<(T * SI) / 256, blk256, 0, stream>>>(
      gs, us, acts, (long long)T * SI);
  wmma_gemm_kernel<<<dim3(cdiv(Hd,BN), cdiv(T,BM), 1), blk128, 0, stream>>>(
      acts, SI, 0, sd_w, Hd, 0, 1, 1, shres, Hd, 0, T, Hd, SI,
      MODE_RESID, h1, Hd, 0.f, nullptr, nullptr, nullptr, nullptr);
  // 13) combine
  combine_kernel<<<(T * Hd) / 256, blk256, 0, stream>>>(shres, slotbf, tkW, out);
}
